// Backbone_78606491452408
// MI455X (gfx1250) — matrix operand order repaired and checked by reading, not compiled, not run on hardware
//
#include <hip/hip_runtime.h>

#define HIDDEN 128
#define IN_DIM 5
#define NEG_SLOPE 0.01f

typedef __attribute__((ext_vector_type(16))) __bf16        v16bf;
typedef __attribute__((ext_vector_type(8)))  float         v8f;
typedef __attribute__((ext_vector_type(4)))  unsigned int  u32x4;

union Frag16 { v16bf bf; u32x4 q[2]; };

__device__ __forceinline__ unsigned short f2bf(float f) {
    unsigned int u = __float_as_uint(f);
    u += 0x7FFFu + ((u >> 16) & 1u);          // round-to-nearest-even
    return (unsigned short)(u >> 16);
}
__device__ __forceinline__ unsigned int pack_bf2(float a, float b) {
    return (unsigned int)f2bf(a) | ((unsigned int)f2bf(b) << 16);
}
__device__ __forceinline__ float leaky(float v) { return v > 0.f ? v : NEG_SLOPE * v; }
__device__ __forceinline__ void atomAddF(float* p, float v) {
    (void)__hip_atomic_fetch_add(p, v, __ATOMIC_RELAXED, __HIP_MEMORY_SCOPE_AGENT);
}

// ---------------- zero fill (vectorized) ----------------
__global__ void zero_f32v4(float* __restrict__ p, int n4) {
    int t = blockIdx.x * blockDim.x + threadIdx.x;
    if (t < n4) {
        u32x4 z = {0u, 0u, 0u, 0u};
        *(u32x4*)(p + 4 * (size_t)t) = z;
    }
}

// ---------------- W[128x128] f32 row-major -> Wt[n][k] bf16 ----------------
__global__ void w_to_bf16_t(const float* __restrict__ W, unsigned short* __restrict__ Wt) {
    int t = blockIdx.x * blockDim.x + threadIdx.x;   // 16384 threads
    int k = t >> 7, n = t & 127;
    Wt[(size_t)n * 128 + k] = f2bf(W[(size_t)k * 128 + n]);
}

// ---------------- layer 1 edge message + scatter (D=5) ----------------
__global__ void edge_msg_d5(const float* __restrict__ x, const int* __restrict__ ei,
                            const float* __restrict__ ea, const float* __restrict__ We,
                            const float* __restrict__ be, float* __restrict__ agg, int E) {
    int e = blockIdx.x * blockDim.x + threadIdx.x;
    if (e >= E) return;
    int src = ei[e], dst = ei[(size_t)E + e];
    float a0 = ea[2 * (size_t)e], a1 = ea[2 * (size_t)e + 1];
    const float* xp = x + (size_t)src * IN_DIM;
    float* ap = agg + (size_t)dst * IN_DIM;
#pragma unroll
    for (int f = 0; f < IN_DIM; ++f) {
        float m = fmaxf(xp[f] + a0 * We[f] + a1 * We[IN_DIM + f] + be[f], 0.f);
        atomAddF(ap + f, m);
    }
}

// ---------------- layer 1 node projection (D=5 -> 128) ----------------
__global__ void node_lin_d5(const float* __restrict__ x, const float* __restrict__ agg,
                            const float* __restrict__ W, const float* __restrict__ b,
                            float* __restrict__ h, int N) {
    long t = (long)blockIdx.x * blockDim.x + threadIdx.x;
    int n = (int)(t >> 7);
    if (n >= N) return;
    int f = (int)t & 127;
    float s = b[f];
#pragma unroll
    for (int k = 0; k < IN_DIM; ++k)
        s += (x[(size_t)n * IN_DIM + k] + agg[(size_t)n * IN_DIM + k]) * W[k * HIDDEN + f];
    h[(size_t)n * HIDDEN + f] = leaky(s);
}

// ---------------- edge message + scatter for D=128 (one wave32 per edge) ----------------
__global__ void edge_msg_128(const float* __restrict__ h, const int* __restrict__ ei,
                             const float* __restrict__ ea, const float* __restrict__ We,
                             const float* __restrict__ be, float* __restrict__ agg, int E) {
    long t = (long)blockIdx.x * blockDim.x + threadIdx.x;
    int e = (int)(t >> 5);
    if (e >= E) return;
    int f = ((int)t & 31) * 4;
    int src = ei[e];
    int dst = ei[(size_t)E + e];
    float a0 = ea[2 * (size_t)e], a1 = ea[2 * (size_t)e + 1];
    const float4 w0 = *(const float4*)(We + f);
    const float4 w1 = *(const float4*)(We + HIDDEN + f);
    const float4 bb = *(const float4*)(be + f);
    const float4 xv = *(const float4*)(h + (size_t)src * HIDDEN + f);
    float m0 = fmaxf(xv.x + a0 * w0.x + a1 * w1.x + bb.x, 0.f);
    float m1 = fmaxf(xv.y + a0 * w0.y + a1 * w1.y + bb.y, 0.f);
    float m2 = fmaxf(xv.z + a0 * w0.z + a1 * w1.z + bb.z, 0.f);
    float m3 = fmaxf(xv.w + a0 * w0.w + a1 * w1.w + bb.w, 0.f);
    float* ap = agg + (size_t)dst * HIDDEN + f;   // agg is 51MB: L2-resident atomics
    atomAddF(ap + 0, m0);
    atomAddF(ap + 1, m1);
    atomAddF(ap + 2, m2);
    atomAddF(ap + 3, m3);
}

// ---------------- node projection via bf16 WMMA: hout = leaky((hin+agg) @ W + b) ----------------
// Wt: bf16, transposed [N=128][K=128]. Each block: 16-node M-tile, 8 waves x 16-col N-tiles.
__global__ void node_gemm_wmma(const float* __restrict__ hin, const float* __restrict__ agg,
                               const unsigned short* __restrict__ Wt,
                               const float* __restrict__ bias, float* __restrict__ hout,
                               int nTiles, int N) {
    __shared__ __align__(16) unsigned short in_s[16][136];   // 16 rows x 128 bf16, padded stride
    const int tid  = threadIdx.x;
    const int wid  = tid >> 5;
    const int lane = tid & 31;
    const int nlo  = lane & 15;
    const int hi   = lane >> 4;
    const int ncol = wid * 16 + nlo;

    // B fragments for all 4 K-steps, held in registers for the whole kernel.
    // 16-bit B (KxN): N=lane&15; elems 0..7 -> K=32s+8*hi+0..7, elems 8..15 -> +16
    Frag16 bfrag[4];
#pragma unroll
    for (int s = 0; s < 4; ++s) {
        const unsigned short* p = Wt + (size_t)ncol * 128 + s * 32 + hi * 8;
        bfrag[s].q[0] = *(const u32x4*)(p);
        bfrag[s].q[1] = *(const u32x4*)(p + 16);
    }

    const float bv = bias[ncol];       // loop-invariant: hoisted
    const int srow = tid >> 4;         // staging: row 0..15
    const int scol = (tid & 15) * 8;   // staging: col 0..120 step 8
    const int mstep = gridDim.x * 16;

    for (int t = blockIdx.x; t < nTiles; t += gridDim.x) {
        const int m0 = t * 16;
        {   // stage (hin + agg) tile as bf16 into LDS
            int r = m0 + srow; if (r >= N) r = N - 1;
            const float* hp = hin + (size_t)r * 128 + scol;
            const float* ap = agg + (size_t)r * 128 + scol;
            u32x4 v;
            v.x = pack_bf2(hp[0] + ap[0], hp[1] + ap[1]);
            v.y = pack_bf2(hp[2] + ap[2], hp[3] + ap[3]);
            v.z = pack_bf2(hp[4] + ap[4], hp[5] + ap[5]);
            v.w = pack_bf2(hp[6] + ap[6], hp[7] + ap[7]);
            *(u32x4*)&in_s[srow][scol] = v;
        }
        // prefetch next tile's staging rows while this tile computes
        {
            int rn = m0 + mstep + srow;
            if (rn < N) {
                __builtin_prefetch(hin + (size_t)rn * 128 + scol, 0, 3);
                __builtin_prefetch(agg + (size_t)rn * 128 + scol, 0, 3);
            }
        }
        __syncthreads();

        v8f acc = {0.f, 0.f, 0.f, 0.f, 0.f, 0.f, 0.f, 0.f};
#pragma unroll
        for (int s = 0; s < 4; ++s) {
            // 16-bit A (MxK): lane row M=nlo; elems 0..7 -> K=32s+8*hi+0..7, elems 8..15 -> +16
            Frag16 a;
            const unsigned short* p = &in_s[nlo][s * 32 + hi * 8];
            a.q[0] = *(const u32x4*)(p);
            a.q[1] = *(const u32x4*)(p + 16);
            acc = __builtin_amdgcn_wmma_f32_16x16x32_bf16(false, a.bf, false, bfrag[s].bf,
                                                          (short)0, acc, false, false);
        }

        // C/D layout: VGPR r -> M = r + 8*hi, N = lane&15
        const int mbase = m0 + hi * 8;
        if (m0 + 16 <= N) {
            // full tile: uniform branch, unguarded stores
#pragma unroll
            for (int r = 0; r < 8; ++r)
                hout[(size_t)(mbase + r) * 128 + ncol] = leaky(acc[r] + bv);
        } else {
#pragma unroll
            for (int r = 0; r < 8; ++r) {
                int row = mbase + r;
                if (row < N) hout[(size_t)row * 128 + ncol] = leaky(acc[r] + bv);
            }
        }
        __syncthreads();   // before next tile re-stages in_s
    }
}

extern "C" void kernel_launch(void* const* d_in, const int* in_sizes, int n_in,
                              void* d_out, int out_size, void* d_ws, size_t ws_size,
                              hipStream_t stream) {
    (void)n_in; (void)out_size; (void)ws_size;
    const float* x   = (const float*)d_in[0];
    const int*   ei  = (const int*)d_in[1];
    const float* ea  = (const float*)d_in[2];
    const float* W0  = (const float*)d_in[3];
    const float* b0  = (const float*)d_in[4];
    const float* We0 = (const float*)d_in[5];
    const float* be0 = (const float*)d_in[6];
    const float* W1  = (const float*)d_in[7];
    const float* b1  = (const float*)d_in[8];
    const float* We1 = (const float*)d_in[9];
    const float* be1 = (const float*)d_in[10];
    const float* W2  = (const float*)d_in[11];
    const float* b2  = (const float*)d_in[12];
    const float* We2 = (const float*)d_in[13];
    const float* be2 = (const float*)d_in[14];

    const int N = in_sizes[0] / IN_DIM;
    const int E = in_sizes[2] / 2;

    float* h   = (float*)d_ws;                               // [N,128] f32
    float* agg = h + (size_t)N * HIDDEN;                     // [N,128] f32
    unsigned short* Wt1 = (unsigned short*)(agg + (size_t)N * HIDDEN);  // [128,128] bf16
    unsigned short* Wt2 = Wt1 + HIDDEN * HIDDEN;
    float* out = (float*)d_out;

    const int nTiles  = (N + 15) / 16;
    const int n4full  = (N * HIDDEN) / 4;
    long ethreads     = (long)E * 32;
    const int eblocks = (int)((ethreads + 255) / 256);

    // weights -> bf16 transposed
    w_to_bf16_t<<<64, 256, 0, stream>>>(W1, Wt1);
    w_to_bf16_t<<<64, 256, 0, stream>>>(W2, Wt2);

    // ---- layer 1 (D = 5) ----
    {
        int n4 = (N * IN_DIM + 3) / 4;                       // agg buffer big enough for overshoot
        zero_f32v4<<<(n4 + 255) / 256, 256, 0, stream>>>(agg, n4);
        edge_msg_d5<<<(E + 255) / 256, 256, 0, stream>>>(x, ei, ea, We0, be0, agg, E);
        long tot = (long)N * HIDDEN;
        node_lin_d5<<<(int)((tot + 255) / 256), 256, 0, stream>>>(x, agg, W0, b0, h, N);
    }
    // ---- layer 2 (D = 128), in-place on h ----
    zero_f32v4<<<(n4full + 255) / 256, 256, 0, stream>>>(agg, n4full);
    edge_msg_128<<<eblocks, 256, 0, stream>>>(h, ei, ea, We1, be1, agg, E);
    node_gemm_wmma<<<1250, 256, 0, stream>>>(h, agg, Wt1, b1, h, nTiles, N);

    // ---- layer 3 (D = 128), writes d_out ----
    zero_f32v4<<<(n4full + 255) / 256, 256, 0, stream>>>(agg, n4full);
    edge_msg_128<<<eblocks, 256, 0, stream>>>(h, ei, ea, We2, be2, agg, E);
    node_gemm_wmma<<<1250, 256, 0, stream>>>(h, agg, Wt2, b2, out, nTiles, N);
}
